// DN_21758304321893
// MI455X (gfx1250) — matrix operand order repaired and checked by reading, not compiled, not run on hardware
//
#include <hip/hip_runtime.h>
#include <hip/hip_bf16.h>

// ---------------------------------------------------------------------------
// Problem constants (fixed by the reference)
// ---------------------------------------------------------------------------
constexpr int nB = 1024;   // batch
constexpr int nX = 1024;   // x features (32*32)
constexpr int nY = 16384;  // codebook size
constexpr int nZ = 1000;   // classes
#define TB 256

typedef float  f32x8   __attribute__((ext_vector_type(8)));
typedef __bf16 bf16x8  __attribute__((ext_vector_type(8)));
typedef __bf16 bf16x16 __attribute__((ext_vector_type(16)));

// output layout (floats), reference tuple order
constexpr size_t o_act  = 0;
constexpr size_t o_mr   = 1;
constexpr size_t o_Wx   = o_mr + nB;
constexpr size_t o_Wz   = o_Wx + (size_t)nY * nX;
constexpr size_t o_U    = o_Wz + (size_t)nY * nZ;
constexpr size_t o_thr  = o_U  + (size_t)nZ * nY;
constexpr size_t o_age  = o_thr + nY;
constexpr size_t o_zage = o_age + nY;

// ---------------------------------------------------------------------------
// helpers: order-preserving float<->uint key, 64-bit shuffle-xor
// ---------------------------------------------------------------------------
__device__ __forceinline__ unsigned fkey(float v) {
  unsigned u = __float_as_uint(v);
  return (u & 0x80000000u) ? ~u : (u | 0x80000000u);
}
__device__ __forceinline__ float fdec(unsigned k) {
  unsigned u = (k & 0x80000000u) ? (k & 0x7FFFFFFFu) : ~k;
  return __uint_as_float(u);
}
__device__ __forceinline__ unsigned long long sx64(unsigned long long v, int m) {
  unsigned lo = __shfl_xor((unsigned)v, m, 32);
  unsigned hi = __shfl_xor((unsigned)(v >> 32), m, 32);
  return ((unsigned long long)hi << 32) | lo;
}

// ---------------------------------------------------------------------------
// K0: zero per-call scratch + d_out[0]
// ---------------------------------------------------------------------------
__global__ __launch_bounds__(TB) void k_init(
    unsigned long long* __restrict__ best, unsigned long long* __restrict__ fbest,
    unsigned* __restrict__ cnt, unsigned* __restrict__ segmin,
    unsigned* __restrict__ cz, unsigned* __restrict__ flag, float* __restrict__ outAct) {
  int i = blockIdx.x * blockDim.x + threadIdx.x;
  if (i < nB) { best[i] = 0ull; fbest[i] = 0ull; }
  if (i < nY) { cnt[i] = 0u; segmin[i] = 0xFFFFFFFFu; }
  if (i < nZ) cz[i] = 0u;
  if (i == 0) { *flag = 0u; *outAct = 0.0f; }
}

// K0b: any unactivated y-neuron?
__global__ __launch_bounds__(TB) void k_flag(const float* __restrict__ yAge,
                                             unsigned* __restrict__ flag) {
  int i = blockIdx.x * blockDim.x + threadIdx.x;
  if (i < nY && yAge[i] < 1.0f) atomicOr(flag, 1u);
}

// ---------------------------------------------------------------------------
// K1: fused L2-normalize of x rows -> f32 + bf16 copies (block per row)
// ---------------------------------------------------------------------------
__global__ __launch_bounds__(TB) void k_xnorm(const float* __restrict__ x,
                                              float* __restrict__ xnf,
                                              __bf16* __restrict__ xnb) {
  __shared__ float red[TB];
  __shared__ float sc;
  int b = blockIdx.x;
  const float* p = x + (size_t)b * nX;
  float s = 0.f;
  for (int k = threadIdx.x; k < nX; k += TB) { float v = p[k]; s += v * v; }
  red[threadIdx.x] = s; __syncthreads();
  for (int w = TB / 2; w > 0; w >>= 1) {
    if (threadIdx.x < w) red[threadIdx.x] += red[threadIdx.x + w];
    __syncthreads();
  }
  if (threadIdx.x == 0) sc = 1.0f / fmaxf(sqrtf(red[0]), 1e-12f);
  __syncthreads();
  float s2 = sc;
  for (int k = threadIdx.x; k < nX; k += TB) {
    float v = p[k] * s2;
    xnf[(size_t)b * nX + k] = v;
    xnb[(size_t)b * nX + k] = (__bf16)v;
  }
}

// K2: generic row inverse-norm (block per row)
__global__ __launch_bounds__(TB) void k_rownorm(const float* __restrict__ in,
                                                float* __restrict__ rinv, int cols) {
  __shared__ float red[TB];
  int r = blockIdx.x;
  const float* p = in + (size_t)r * cols;
  float s = 0.f;
  for (int k = threadIdx.x; k < cols; k += TB) { float v = p[k]; s += v * v; }
  red[threadIdx.x] = s; __syncthreads();
  for (int w = TB / 2; w > 0; w >>= 1) {
    if (threadIdx.x < w) red[threadIdx.x] += red[threadIdx.x + w];
    __syncthreads();
  }
  if (threadIdx.x == 0) rinv[r] = 1.0f / fmaxf(sqrtf(red[0]), 1e-12f);
}

// K2b: Wx row inverse-norm + one-time pre-normalized bf16 staging
__global__ __launch_bounds__(TB) void k_wxnorm(const float* __restrict__ in,
                                               float* __restrict__ rinv,
                                               __bf16* __restrict__ outb) {
  __shared__ float red[TB];
  __shared__ float sc;
  int r = blockIdx.x;
  const float* p = in + (size_t)r * nX;
  float s = 0.f;
  for (int k = threadIdx.x; k < nX; k += TB) { float v = p[k]; s += v * v; }
  red[threadIdx.x] = s; __syncthreads();
  for (int w = TB / 2; w > 0; w >>= 1) {
    if (threadIdx.x < w) red[threadIdx.x] += red[threadIdx.x + w];
    __syncthreads();
  }
  if (threadIdx.x == 0) {
    float v = 1.0f / fmaxf(sqrtf(red[0]), 1e-12f);
    rinv[r] = v; sc = v;
  }
  __syncthreads();
  float s2 = sc;
  for (int k = threadIdx.x; k < nX; k += TB)
    outb[(size_t)r * nX + k] = (__bf16)(p[k] * s2);
}

// ---------------------------------------------------------------------------
// K3: bf16 WMMA GEMM  y_pre = 0.5*(xn @ Wxn^T) + 0.5*Wzn[:,z[b]]
//     fused per-b argmax (+ masked fallback argmax) via packed-key atomics.
// Each wave owns a 64(b) x 16(j) tile (MT=4 accumulators, B fragment reused
// across 4 WMMAs). Block = 8 waves = 64(b) x 128(j). Grid: (nY/128, nB/64).
// sched_group_barrier pins each K step to "10 VMEM reads, then 4 WMMAs" so
// all fragments stay live and the waits become progressive (6/4/2/0) instead
// of a full s_wait_loadcnt 0 before every WMMA.
// PRE=true : B fragments are pre-normalized bf16 (two b128 loads, no VALU).
// PRE=false: B fragments converted from raw f32 in-register.
// ---------------------------------------------------------------------------
template <bool PRE>
__global__ __launch_bounds__(TB) void k_gemm_argmax(
    const __bf16* __restrict__ xnb, const float* __restrict__ Wx,
    const __bf16* __restrict__ Wxb, const float* __restrict__ rWx,
    const float* __restrict__ Wz, const float* __restrict__ rWz,
    const float* __restrict__ yAge, const int* __restrict__ zl,
    unsigned long long* __restrict__ best, unsigned long long* __restrict__ fbest) {
  constexpr int MT = 4;                 // batch tiles per wave
  const int lane = threadIdx.x & 31;
  const int wave = threadIdx.x >> 5;
  const int n    = lane & 15;   // N index within 16x16 tile (B & C/D operand lane)
  const int half = lane >> 4;   // K-half / M-half selector
  const int j     = blockIdx.x * 128 + wave * 16 + n;   // codebook neuron
  const int bBase = blockIdx.y * (16 * MT);             // batch tile base

  const __bf16* aBase = xnb + (size_t)(bBase + n) * nX;
  const __bf16* bRowB = Wxb + (size_t)j * nX + half * 16;
  const float*  bRowF = Wx  + (size_t)j * nX + half * 16;
  const float   rw    = PRE ? 1.0f : rWx[j];

  f32x8 acc[MT];
#pragma unroll
  for (int t = 0; t < MT; ++t) acc[t] = (f32x8){0.f,0.f,0.f,0.f,0.f,0.f,0.f,0.f};

  for (int k0 = 0; k0 < nX; k0 += 32) {
    // ---- phase 1: issue all fragment loads for this K step ----
    bf16x8 b0, b1;
    float4 bf[4];
    if (PRE) {
      b0 = *(const bf16x8*)(bRowB + k0);
      b1 = *(const bf16x8*)(bRowB + k0 + 8);
    } else {
      const float4* bp = (const float4*)(bRowF + k0);
#pragma unroll
      for (int i = 0; i < 4; ++i) bf[i] = bp[i];
    }
    bf16x8 a0[MT], a1[MT];
#pragma unroll
    for (int t = 0; t < MT; ++t) {
      const __bf16* aRow = aBase + (size_t)t * 16 * nX;
      a0[t] = *(const bf16x8*)(aRow + k0 + half * 16);
      a1[t] = *(const bf16x8*)(aRow + k0 + half * 16 + 8);
    }
    // ---- phase 2: assemble B fragment, then 4 WMMAs ----
    bf16x16 Bm;
    if (PRE) {
      Bm = __builtin_shufflevector(b0, b1, 0, 1, 2, 3, 4, 5, 6, 7,
                                   8, 9, 10, 11, 12, 13, 14, 15);
    } else {
#pragma unroll
      for (int i = 0; i < 4; ++i) {
        Bm[4 * i + 0] = (__bf16)(bf[i].x * rw);
        Bm[4 * i + 1] = (__bf16)(bf[i].y * rw);
        Bm[4 * i + 2] = (__bf16)(bf[i].z * rw);
        Bm[4 * i + 3] = (__bf16)(bf[i].w * rw);
      }
    }
#pragma unroll
    for (int t = 0; t < MT; ++t) {
      bf16x16 A = __builtin_shufflevector(a0[t], a1[t], 0, 1, 2, 3, 4, 5, 6, 7,
                                          8, 9, 10, 11, 12, 13, 14, 15);
      acc[t] = __builtin_amdgcn_wmma_f32_16x16x32_bf16(false, A, false, Bm,
                                                       (short)0, acc[t], false, false);
    }
    if (PRE) {
      // scheduling groups: 10 VMEM reads (0x0020), then 4 MFMA/WMMA (0x0008)
      __builtin_amdgcn_sched_group_barrier(0x0020, 10, 0);
      __builtin_amdgcn_sched_group_barrier(0x0008, 4, 0);
    }
  }

  // epilogue: y_pre, per-b max + masked-fallback max, packed (key | ~j)
  const float un = (yAge[j] < 1.0f) ? 1.0f : 0.0f;
  const float rz = rWz[j];
  const float* wzRow = Wz + (size_t)j * nZ;
#pragma unroll
  for (int t = 0; t < MT; ++t) {
#pragma unroll
    for (int e = 0; e < 8; ++e) {
      int b  = bBase + t * 16 + half * 8 + e;   // D layout: lanes 16-31 hold M = e+8
      int zb = zl[b];
      float ypre = 0.5f * acc[t][e] + 0.5f * (wzRow[zb] * rz);
      unsigned long long p  = ((unsigned long long)fkey(ypre) << 32) | (unsigned)(~(unsigned)j);
      unsigned long long p2 = ((unsigned long long)fkey(ypre * un) << 32) | (unsigned)(~(unsigned)j);
#pragma unroll
      for (int m = 1; m < 16; m <<= 1) {
        unsigned long long q = sx64(p, m);   if (q > p)   p = q;
        unsigned long long q2 = sx64(p2, m); if (q2 > p2) p2 = q2;
      }
      if (n == 0) {
        atomicMax(&best[b], p);
        atomicMax(&fbest[b], p2);
      }
    }
  }
}

// ---------------------------------------------------------------------------
// K4: decide winners, emit max_resp, cnt/segmin/cz histograms
// ---------------------------------------------------------------------------
__global__ __launch_bounds__(TB) void k_decide(
    const unsigned long long* __restrict__ best,
    const unsigned long long* __restrict__ fbest,
    const float* __restrict__ yAge, const float* __restrict__ thrOld,
    const int* __restrict__ zl, const unsigned* __restrict__ flag,
    int* __restrict__ win, unsigned* __restrict__ cnt,
    unsigned* __restrict__ segmin, unsigned* __restrict__ cz,
    float* __restrict__ outMR) {
  int b = blockIdx.x * blockDim.x + threadIdx.x;
  if (b >= nB) return;
  unsigned long long p = best[b];
  unsigned key = (unsigned)(p >> 32);
  int j  = (int)(~(unsigned)(p & 0xFFFFFFFFull));
  int fj = (int)(~(unsigned)(fbest[b] & 0xFFFFFFFFull));
  float v = fdec(key);
  outMR[b] = v;
  bool judge = (v > thrOld[j]) || (yAge[j] < 1.0f);
  bool anyUn = (*flag) != 0u;
  int w = (judge || !anyUn) ? j : fj;
  win[b] = w;
  atomicAdd(&cnt[w], 1u);
  atomicMin(&segmin[w], key);       // ordered key => float min
  atomicAdd(&cz[zl[b]], 1u);
}

// ---------------------------------------------------------------------------
// K5: per-Y finalize (scale factors, age_new, thr_new, activated count)
// ---------------------------------------------------------------------------
__global__ __launch_bounds__(TB) void k_finY(
    const float* __restrict__ yAge, const float* __restrict__ thrOld,
    const unsigned* __restrict__ cnt, const unsigned* __restrict__ segmin,
    float* __restrict__ sW, float* __restrict__ aW,
    float* __restrict__ outThr, float* __restrict__ outAge,
    float* __restrict__ outAct) {
  int j = blockIdx.x * blockDim.x + threadIdx.x;
  if (j >= nY) return;
  float age = yAge[j];
  float c   = (float)cnt[j];
  float lr  = 1.0f / (age + 1.0f);
  float div = (c > 0.f) ? c : 1.0f;
  sW[j] = 1.0f - lr * c / div;   // dense row scale for Wx/Wz
  aW[j] = lr / div;              // scatter scale
  float ageN = age + c;
  outAge[j] = ageN;
  if (ageN >= 1.0f) atomicAdd(outAct, 1.0f);
  float sm   = fdec(segmin[j]);          // +inf-ish when no winner
  float ycur = fminf(sm, 3.0f);
  ycur = (ycur > 2.0f) ? 0.0f : ycur;
  float lre = (c > 0.f) ? lr : 0.0f;
  outThr[j] = lre * ycur + (1.0f - lre) * thrOld[j];
}

// K5b: per-Z finalize
__global__ __launch_bounds__(TB) void k_finZ(
    const float* __restrict__ zAge, const unsigned* __restrict__ cz,
    float* __restrict__ sU, float* __restrict__ aU, float* __restrict__ outZAge) {
  int c = blockIdx.x * blockDim.x + threadIdx.x;
  if (c >= nZ) return;
  float age = zAge[c];
  float n   = (float)cz[c];
  float zlr = 1.0f / (age + 1.0f);
  float dz  = (n > 0.f) ? n : 1.0f;
  sU[c] = 1.0f - zlr * n / dz;
  aU[c] = zlr / dz;
  outZAge[c] = age + n;
}

// ---------------------------------------------------------------------------
// K6: fused normalize+decay write: out[r,k] = in[r,k] * rnorm[r] * fac[r]
// grid = (colChunks, rows)
// ---------------------------------------------------------------------------
__global__ __launch_bounds__(TB) void k_scale_write(
    const float* __restrict__ in, float* __restrict__ out,
    const float* __restrict__ rn, const float* __restrict__ fac, int cols) {
  int r = blockIdx.y;
  float f = rn[r] * fac[r];
  const float* ip = in + (size_t)r * cols;
  float* op = out + (size_t)r * cols;
  for (int k = blockIdx.x * blockDim.x + threadIdx.x; k < cols;
       k += gridDim.x * blockDim.x)
    op[k] = ip[k] * f;
}

// ---------------------------------------------------------------------------
// K7: sparse scatter updates (B winners)
// ---------------------------------------------------------------------------
__global__ __launch_bounds__(TB) void k_scatter_x(
    const int* __restrict__ win, const float* __restrict__ aW,
    const float* __restrict__ xnf, float* __restrict__ WxOut) {
  int b = blockIdx.x;
  int w = win[b];
  float a = aW[w];
  const float* xr = xnf + (size_t)b * nX;
  float* orow = WxOut + (size_t)w * nX;
  for (int k = threadIdx.x; k < nX; k += blockDim.x)
    atomicAdd(&orow[k], a * xr[k]);
}

__global__ __launch_bounds__(TB) void k_scatter_small(
    const int* __restrict__ win, const int* __restrict__ zl,
    const float* __restrict__ aW, const float* __restrict__ aU,
    float* __restrict__ WzOut, float* __restrict__ UOut) {
  int b = blockIdx.x * blockDim.x + threadIdx.x;
  if (b >= nB) return;
  int w = win[b];
  int c = zl[b];
  atomicAdd(&WzOut[(size_t)w * nZ + c], aW[w]);
  atomicAdd(&UOut[(size_t)c * nY + w], aU[c]);
}

// ---------------------------------------------------------------------------
// launcher
// ---------------------------------------------------------------------------
extern "C" void kernel_launch(void* const* d_in, const int* in_sizes, int n_in,
                              void* d_out, int out_size, void* d_ws, size_t ws_size,
                              hipStream_t stream) {
  (void)in_sizes; (void)n_in; (void)out_size;
  const float* x      = (const float*)d_in[0];
  const int*   z      = (const int*)d_in[1];
  const float* x2y_w  = (const float*)d_in[2];
  const float* z2y_w  = (const float*)d_in[3];
  const float* y2z_w  = (const float*)d_in[4];
  const float* y_age  = (const float*)d_in[5];
  const float* z_age  = (const float*)d_in[6];
  const float* y_thr  = (const float*)d_in[7];
  float* out = (float*)d_out;

  // workspace carving (256B aligned slots; ~6.5 MB core + optional 32 MB bf16 Wx)
  char* W = (char*)d_ws;
  auto carve = [&](size_t bytes) {
    void* p = (void*)W;
    W += (bytes + 255) & ~(size_t)255;
    return p;
  };
  float*              xnf    = (float*)carve((size_t)nB * nX * 4);
  __bf16*             xnb    = (__bf16*)carve((size_t)nB * nX * 2);
  unsigned long long* best   = (unsigned long long*)carve((size_t)nB * 8);
  unsigned long long* fbest  = (unsigned long long*)carve((size_t)nB * 8);
  float*              rWx    = (float*)carve((size_t)nY * 4);
  float*              rWz    = (float*)carve((size_t)nY * 4);
  float*              rU     = (float*)carve((size_t)nZ * 4);
  unsigned*           cnt    = (unsigned*)carve((size_t)nY * 4);
  unsigned*           segmin = (unsigned*)carve((size_t)nY * 4);
  int*                win    = (int*)carve((size_t)nB * 4);
  unsigned*           cz     = (unsigned*)carve((size_t)nZ * 4);
  float*              sW     = (float*)carve((size_t)nY * 4);
  float*              aW     = (float*)carve((size_t)nY * 4);
  float*              sU     = (float*)carve((size_t)nZ * 4);
  float*              aU     = (float*)carve((size_t)nZ * 4);
  unsigned*           flag   = (unsigned*)carve(256);
  __bf16*             Wxb    = (__bf16*)carve((size_t)nY * nX * 2);  // optional staging
  const bool pre = ((size_t)(W - (char*)d_ws) <= ws_size);           // fits incl. Wxb?

  // 0) init scratch + activated-count output
  k_init<<<(nY + TB - 1) / TB, TB, 0, stream>>>(best, fbest, cnt, segmin, cz,
                                                flag, out + o_act);
  k_flag<<<(nY + TB - 1) / TB, TB, 0, stream>>>(y_age, flag);

  // 1) normalizations (x fused f32+bf16; Wx optionally staged to bf16 once)
  k_xnorm<<<nB, TB, 0, stream>>>(x, xnf, xnb);
  if (pre)
    k_wxnorm<<<nY, TB, 0, stream>>>(x2y_w, rWx, Wxb);
  else
    k_rownorm<<<nY, TB, 0, stream>>>(x2y_w, rWx, nX);
  k_rownorm<<<nY, TB, 0, stream>>>(z2y_w, rWz, nZ);
  k_rownorm<<<nZ, TB, 0, stream>>>(y2z_w, rU, nY);

  // 2) WMMA GEMM + fused argmax competition (64b x 128j per block)
  dim3 gg(nY / 128, nB / 64);
  if (pre)
    k_gemm_argmax<true><<<gg, TB, 0, stream>>>(xnb, x2y_w, Wxb, rWx, z2y_w,
                                               rWz, y_age, z, best, fbest);
  else
    k_gemm_argmax<false><<<gg, TB, 0, stream>>>(xnb, x2y_w, xnb, rWx, z2y_w,
                                                rWz, y_age, z, best, fbest);

  // 3) winners, histograms, per-row factors
  k_decide<<<(nB + TB - 1) / TB, TB, 0, stream>>>(best, fbest, y_age, y_thr, z,
                                                  flag, win, cnt, segmin, cz,
                                                  out + o_mr);
  k_finY<<<(nY + TB - 1) / TB, TB, 0, stream>>>(y_age, y_thr, cnt, segmin, sW,
                                                aW, out + o_thr, out + o_age,
                                                out + o_act);
  k_finZ<<<(nZ + TB - 1) / TB, TB, 0, stream>>>(z_age, cz, sU, aU, out + o_zage);

  // 4) dense single-pass weight writes: raw * rnorm * decay
  k_scale_write<<<dim3(1, nY), TB, 0, stream>>>(x2y_w, out + o_Wx, rWx, sW, nX);
  k_scale_write<<<dim3(1, nY), TB, 0, stream>>>(z2y_w, out + o_Wz, rWz, sW, nZ);
  k_scale_write<<<dim3(16, nZ), TB, 0, stream>>>(y2z_w, out + o_U, rU, sU, nY);

  // 5) sparse Hebbian scatter (B=1024 winners)
  k_scatter_x<<<nB, TB, 0, stream>>>(win, aW, xnf, out + o_Wx);
  k_scatter_small<<<(nB + TB - 1) / TB, TB, 0, stream>>>(win, z, aW, aU,
                                                         out + o_Wz, out + o_U);
}
